// ClassificationByRetrieval_86955907875298
// MI455X (gfx1250) — matrix operand order repaired and checked by reading, not compiled, not run on hardware
//
#include <hip/hip_runtime.h>
#include <cstdint>
#include <cstddef>

// Problem constants (match reference)
#define B_Q    256
#define N_IDX  200000
#define D_DIM  128
#define C_CLS  1000

// Tiling
#define NSLABS 16            // N split into 16 slabs (grid.x)
#define MT     16            // query rows per workgroup (grid.y = B_Q/MT = 16)
#define NTILE  256           // index columns per buffered tile (8 waves x 2 subtiles x 16)
#define LDS_BSTRIDE 136      // bf16 elems per LDS row: 128 data + 8 pad (TDM pad 4 DWORDs)
#define TILE_BYTES (NTILE * LDS_BSTRIDE * 2)

typedef __attribute__((ext_vector_type(16))) __bf16 v16bf;
typedef __attribute__((ext_vector_type(8)))  float  v8f;
typedef __attribute__((ext_vector_type(4)))  unsigned int u32x4;
typedef __attribute__((ext_vector_type(8)))  int    i32x8;
typedef __attribute__((ext_vector_type(4)))  int    i32x4;
typedef __attribute__((ext_vector_type(4)))  float  f32x4;
typedef __attribute__((ext_vector_type(4)))  unsigned short u16x4;

union Frag16 { u32x4 u[2]; v16bf v; };

#define ORD_SENTINEL ((int)0x80000000)

// Monotonic order-preserving float->int map (max over ints == max over floats)
__device__ __forceinline__ int order_f32(float f) {
    int i = __float_as_int(f);
    return (i >= 0) ? i : (i ^ 0x7fffffff);
}
__device__ __forceinline__ float unorder_f32(int m) {
    int bits = (m >= 0) ? m : (m ^ 0x7fffffff);
    return __int_as_float(bits);
}
// f32 -> bf16 round-to-nearest-even (bit pattern as ushort)
__device__ __forceinline__ unsigned short f32_to_bf16(float f) {
    unsigned int u = __float_as_uint(f);
    u += 0x7fffu + ((u >> 16) & 1u);
    return (unsigned short)(u >> 16);
}

// ---------------------------------------------------------------------------
// TDM: DMA a (up to) NTILE x 128 bf16 tile from idxb[cbase..] into LDS with
// hardware row padding (64 data DWORDs + 4 pad DWORDs -> LDS_BSTRIDE).
// Issued by one wave; completion tracked with TENSORcnt.
// ---------------------------------------------------------------------------
__device__ __forceinline__ void tdm_load_tile(const unsigned short* gbase,
                                              int cbase, int slabEnd,
                                              unsigned ldsAddr) {
    const unsigned long long ga =
        (unsigned long long)(uintptr_t)(gbase + (size_t)cbase * D_DIM);
    const int rem = slabEnd - cbase;     // tensor_dim1: OOB rows read as zero

    u32x4 g0;
    g0[0] = 1u;                                        // count=1 valid descriptor
    g0[1] = ldsAddr;                                   // lds_addr (byte offset)
    g0[2] = (unsigned)(ga & 0xffffffffu);              // global_addr[31:0]
    g0[3] = (unsigned)((ga >> 32) & 0x01ffffffu)       // global_addr[56:32]
          | (2u << 30);                                // type = 2 ("image")

    i32x8 g1;
    g1[0] = (1 << 16)        // data_size = 2 bytes
          | (1 << 20)        // pad_enable
          | (5 << 22)        // pad_interval: 64 DWORDs (=one 128-bf16 row)
          | (3 << 25);       // pad_amount: 4 DWORDs (=8 bf16)
    g1[1] = (D_DIM & 0xffff) << 16;                    // tensor_dim0[15:0]
    g1[2] = ((D_DIM >> 16) & 0xffff)                   // tensor_dim0[31:16]
          | ((rem & 0xffff) << 16);                    // tensor_dim1[15:0]
    g1[3] = ((rem >> 16) & 0xffff)                     // tensor_dim1[31:16]
          | (D_DIM << 16);                             // tile_dim0 = 128
    g1[4] = NTILE;                                     // tile_dim1 = 256, tile_dim2 = 0
    g1[5] = D_DIM;                                     // tensor_dim0_stride[31:0] = 128
    g1[6] = 0;                                         // stride hi + dim1_stride lo
    g1[7] = 0;                                         // dim1_stride hi (2D: unused)

    const i32x4 z4 = {0, 0, 0, 0};
    const i32x8 z8 = {0, 0, 0, 0, 0, 0, 0, 0};
    // 6-arg variant (clang-23 / therock-10.0 headers): groups 2/3 unused (2D).
    __builtin_amdgcn_tensor_load_to_lds(g0, g1, z4, z4, z8, 0);
}

// ---------------------------------------------------------------------------
// Kernel 1: L2-normalize queries + index embeddings, emit bf16.
// One wave (32 lanes) per row of 128 floats (4 per lane).
// ---------------------------------------------------------------------------
__global__ void cbr_normalize_kernel(const float* __restrict__ x,
                                     const float* __restrict__ idx_emb,
                                     unsigned short* __restrict__ xb,
                                     unsigned short* __restrict__ idxb) {
    const int wave = (blockIdx.x * blockDim.x + threadIdx.x) >> 5;
    const int lane = threadIdx.x & 31;
    const int totalRows = N_IDX + B_Q;
    if (wave >= totalRows) return;

    const float* src;
    unsigned short* dst;
    if (wave < N_IDX) {
        src = idx_emb + (size_t)wave * D_DIM;
        dst = idxb    + (size_t)wave * D_DIM;
    } else {
        const int r = wave - N_IDX;
        src = x  + (size_t)r * D_DIM;
        dst = xb + (size_t)r * D_DIM;
    }

    f32x4 v = ((const f32x4*)src)[lane];
    float ss = v.x * v.x + v.y * v.y + v.z * v.z + v.w * v.w;
    #pragma unroll
    for (int off = 16; off > 0; off >>= 1) ss += __shfl_xor(ss, off, 32);
    const float scale = 1.0f / fmaxf(sqrtf(ss), 1e-12f);

    u16x4 o;
    o.x = f32_to_bf16(v.x * scale);
    o.y = f32_to_bf16(v.y * scale);
    o.z = f32_to_bf16(v.z * scale);
    o.w = f32_to_bf16(v.w * scale);
    ((u16x4*)dst)[lane] = o;
}

// ---------------------------------------------------------------------------
// Kernel 2: bf16 WMMA GEMM fused with per-class max.
// grid = (NSLABS, B_Q/MT). Block = 256 threads = 8 waves.
// Double-buffered TDM pipeline: wave 0 DMAs tile t+1 while all waves run
// 8 WMMAs (2 independent accumulator chains) on tile t, then lane-mapped
// ds_max_i32 atomics into the ordered-int class-max table in LDS.
// ---------------------------------------------------------------------------
__global__ void cbr_gemm_segmax_kernel(const unsigned short* __restrict__ xb,
                                       const unsigned short* __restrict__ idxb,
                                       const int* __restrict__ labels,
                                       int* __restrict__ partials) {
    extern __shared__ char smem[];
    int* accI = (int*)(smem + 2 * TILE_BYTES);          // MT x C_CLS ordered-int

    const int slab  = blockIdx.x;
    const int mtile = blockIdx.y;
    const int tid   = threadIdx.x;
    const int lane  = tid & 31;
    const int wv    = tid >> 5;

    // init class-max accumulator
    for (int i = tid; i < MT * C_CLS; i += 256) accI[i] = ORD_SENTINEL;

    // A fragments: constant for whole slab. 16-bit A 16x32 layout:
    // lanes 0-15 (M=lane): K = {kk*32 + 0..7, kk*32 + 16..23}
    // lanes 16-31(M=lane-16): K = {kk*32 + 8..15, kk*32 + 24..31}
    const int m0 = mtile * MT;
    Frag16 afr[4];
    {
        const unsigned short* arow = xb + (size_t)(m0 + (lane & 15)) * D_DIM;
        #pragma unroll
        for (int kk = 0; kk < 4; ++kk) {
            const int kbase = kk * 32 + ((lane >> 4) << 3);
            afr[kk].u[0] = *(const u32x4*)(arow + kbase);
            afr[kk].u[1] = *(const u32x4*)(arow + kbase + 16);
        }
    }

    const int slabBeg = slab * (N_IDX / NSLABS);
    const int slabEnd = (slab == NSLABS - 1) ? N_IDX : slabBeg + (N_IDX / NSLABS);

    // LDS byte address of buffer 0 (low 32 bits of the flat address == LDS offset)
    const unsigned ldsBase = (unsigned)(uintptr_t)(void*)smem;

    // Preload tile 0 into buffer 0.
    if (wv == 0)
        tdm_load_tile(idxb, slabBeg, slabEnd, ldsBase);

    int parity = 0;
    for (int cbase = slabBeg; cbase < slabEnd; cbase += NTILE) {
        const unsigned short* cur =
            (const unsigned short*)(smem + parity * TILE_BYTES);
        const bool hasNext = (cbase + NTILE) < slabEnd;

        if (wv == 0) {
            if (hasNext) {
                tdm_load_tile(idxb, cbase + NTILE, slabEnd,
                              ldsBase + (unsigned)((parity ^ 1) * TILE_BYTES));
                __builtin_amdgcn_s_wait_tensorcnt(1);  // cur done, next in flight
            } else {
                __builtin_amdgcn_s_wait_tensorcnt(0);
            }
        }
        __syncthreads();   // cur tile visible to all waves (and accI init, iter 0)

        // Two independent 16x16 accumulator chains per wave, K=128 (4 steps).
        v8f acc0 = {};
        v8f acc1 = {};
        #pragma unroll
        for (int kk = 0; kk < 4; ++kk) {
            // B 32x16 bf16: lanes 0-15 hold K=kk*32+{0..7,16..23} (N=lane),
            // lanes 16-31 hold K=kk*32+{8..15,24..31}.
            const int koff = kk * 32 + ((lane >> 4) << 3);
            const unsigned short* br0 =
                cur + (wv * 16 + (lane & 15)) * LDS_BSTRIDE + koff;
            const unsigned short* br1 = br0 + 128 * LDS_BSTRIDE;
            Frag16 b0, b1;
            b0.u[0] = *(const u32x4*)(br0);
            b0.u[1] = *(const u32x4*)(br0 + 16);
            b1.u[0] = *(const u32x4*)(br1);
            b1.u[1] = *(const u32x4*)(br1 + 16);
            acc0 = __builtin_amdgcn_wmma_f32_16x16x32_bf16(
                       false, afr[kk].v, false, b0.v, (short)0, acc0, false, false);
            acc1 = __builtin_amdgcn_wmma_f32_16x16x32_bf16(
                       false, afr[kk].v, false, b1.v, (short)0, acc1, false, false);
        }

        // Segment-max: lane owns cols (cbase + j*128 + wv*16 + lane&15),
        // rows (lane>>4)*8 + r.
        const int colLo = cbase + wv * 16 + (lane & 15);
        int* rowBase = accI + ((lane >> 4) << 3) * C_CLS;
        if (colLo < slabEnd) {
            const int cls = labels[colLo];
            #pragma unroll
            for (int r = 0; r < 8; ++r)
                atomicMax(rowBase + r * C_CLS + cls, order_f32(acc0[r]));
        }
        const int colHi = colLo + 128;
        if (colHi < slabEnd) {
            const int cls = labels[colHi];
            #pragma unroll
            for (int r = 0; r < 8; ++r)
                atomicMax(rowBase + r * C_CLS + cls, order_f32(acc1[r]));
        }
        __syncthreads();   // all reads of cur done before it becomes a DMA target
        parity ^= 1;
    }

    // Write this WG's partial class-max table: partials[slab][mtile][MT][C_CLS]
    int* out = partials + (size_t)(slab * (B_Q / MT) + mtile) * (MT * C_CLS);
    for (int i = tid; i < MT * C_CLS; i += 256) out[i] = accI[i];
}

// ---------------------------------------------------------------------------
// Kernel 3: reduce 16 slab partials -> f32 logits [B_Q, C_CLS]
// ---------------------------------------------------------------------------
__global__ void cbr_reduce_kernel(const int* __restrict__ partials,
                                  float* __restrict__ out) {
    const int idx = blockIdx.x * blockDim.x + threadIdx.x;
    if (idx >= B_Q * C_CLS) return;
    const int row = idx / C_CLS;
    const int cls = idx - row * C_CLS;
    const int mtile = row >> 4;
    const int rl    = row & 15;
    int m = ORD_SENTINEL;
    #pragma unroll
    for (int s = 0; s < NSLABS; ++s) {
        const int v = partials[(size_t)(s * (B_Q / MT) + mtile) * (MT * C_CLS)
                               + rl * C_CLS + cls];
        m = (v > m) ? v : m;
    }
    out[idx] = (m == ORD_SENTINEL) ? -3.402823466e38f : unorder_f32(m);
}

// ---------------------------------------------------------------------------
extern "C" void kernel_launch(void* const* d_in, const int* in_sizes, int n_in,
                              void* d_out, int out_size, void* d_ws, size_t ws_size,
                              hipStream_t stream) {
    const float* x       = (const float*)d_in[0];
    const float* idx_emb = (const float*)d_in[1];
    const int*   labels  = (const int*)d_in[2];
    // d_in[3] = num_classes scalar; fixed to C_CLS here.
    float* out = (float*)d_out;

    char* ws = (char*)d_ws;
    unsigned short* xb   = (unsigned short*)ws;                              // 256*128 bf16 = 64KB
    unsigned short* idxb = (unsigned short*)(ws + 65536);                    // 200000*128 bf16
    int* partials = (int*)(ws + 65536 + (size_t)N_IDX * D_DIM * 2);          // 16*16*16*1000 int

    // 1) normalize -> bf16
    {
        const int totalRows = N_IDX + B_Q;               // one wave per row
        const int rowsPerBlock = 256 / 32;
        const int nblk = (totalRows + rowsPerBlock - 1) / rowsPerBlock;
        cbr_normalize_kernel<<<nblk, 256, 0, stream>>>(x, idx_emb, xb, idxb);
    }

    // 2) WMMA GEMM + fused per-class LDS max (double-buffered TDM pipeline)
    {
        const size_t smemBytes = 2 * (size_t)TILE_BYTES + (size_t)MT * C_CLS * 4; // ~198.5KB
        dim3 grid(NSLABS, B_Q / MT);
        cbr_gemm_segmax_kernel<<<grid, 256, smemBytes, stream>>>(xb, idxb, labels, partials);
    }

    // 3) reduce slabs -> logits
    {
        const int total = B_Q * C_CLS;
        cbr_reduce_kernel<<<(total + 255) / 256, 256, 0, stream>>>(partials, out);
    }
}
